// DenseMHA_6373731467597
// MI455X (gfx1250) — matrix operand order repaired and checked by reading, not compiled, not run on hardware
//
#include <hip/hip_runtime.h>
#include <hip/hip_bf16.h>

typedef __attribute__((ext_vector_type(16))) __bf16 v16bf;
typedef __attribute__((ext_vector_type(8)))  float  v8f;
typedef __attribute__((ext_vector_type(4)))  unsigned int u32x4;
typedef __attribute__((ext_vector_type(8)))  int  i32x8;
typedef __attribute__((ext_vector_type(4)))  int  i32x4;

#define HIDDEN_ 2048
#define S_LEN   2048
#define BATCH   2
#define NHEAD   16
#define HDIM    128
#define INNER_  2048
#define MTOT    4096   // BATCH * S_LEN

__device__ __forceinline__ unsigned short f2bf(float f) {
    unsigned int u = __float_as_uint(f);
    u += 0x7fffu + ((u >> 16) & 1u);   // round-to-nearest-even
    return (unsigned short)(u >> 16);
}

union Frag16 {
    uint4 q[2];
    unsigned short h[16];
    v16bf v;
};

// ---------------------------------------------------------------------------
// TDM: async 2D tile load (tile_w x tile_h of 2-byte elements) into LDS.
// Descriptor layout per CDNA5 ISA.
// ---------------------------------------------------------------------------
__device__ __forceinline__ void tdm_load_tile_2d(const unsigned short* gsrc,
                                                 unsigned lds_off,
                                                 unsigned tensor_w, unsigned tensor_h,
                                                 unsigned tile_w, unsigned tile_h,
                                                 unsigned row_stride_elems) {
    unsigned long long ga = (unsigned long long)(uintptr_t)gsrc;
    u32x4 g0;
    g0[0] = 1u;                                             // count=1, user mode
    g0[1] = lds_off;                                        // lds_addr (bytes)
    g0[2] = (unsigned)(ga & 0xFFFFFFFFull);                 // global_addr[31:0]
    g0[3] = (unsigned)((ga >> 32) & 0x01FFFFFFull)          // global_addr[56:32]
          | (2u << 30);                                     // type=2 ("image")
    i32x8 g1;
    g1[0] = (int)(1u << 16);                                // data_size=1 -> 2 bytes
    g1[1] = (int)((tensor_w & 0xFFFFu) << 16);              // tensor_dim0[15:0]
    g1[2] = (int)(((tensor_w >> 16) & 0xFFFFu)
          | ((tensor_h & 0xFFFFu) << 16));                  // dim0 hi | dim1 lo
    g1[3] = (int)(((tensor_h >> 16) & 0xFFFFu)
          | ((tile_w & 0xFFFFu) << 16));                    // dim1 hi | tile_dim0
    g1[4] = (int)(tile_h & 0xFFFFu);                        // tile_dim1 (tile_dim2=0)
    g1[5] = (int)row_stride_elems;                          // tensor_dim0_stride[31:0]
    g1[6] = 0;
    g1[7] = 0;
    i32x4 gz = {0, 0, 0, 0};
#if __clang_major__ >= 23
    i32x8 gz8 = {0, 0, 0, 0, 0, 0, 0, 0};
    __builtin_amdgcn_tensor_load_to_lds(g0, g1, gz, gz, gz8, 0);
#else
    __builtin_amdgcn_tensor_load_to_lds(g0, g1, gz, gz, 0);
#endif
}

// ---------------------------------------------------------------------------
// fp32 -> bf16 convert
// ---------------------------------------------------------------------------
__global__ void f32_to_bf16_kernel(const float* __restrict__ src,
                                   unsigned short* __restrict__ dst, int n) {
    int i = blockIdx.x * blockDim.x + threadIdx.x;
    if (i < n) dst[i] = f2bf(src[i]);
}

// ---------------------------------------------------------------------------
// bf16 GEMM: out[m,n] = sum_k X[m,k]*W[n,k] + bias[n]
//   mode 0: bias + RoPE, write bf16 [B,H,S,D]    (Q,K)
//   mode 1: bias,       write bf16 [B,H,D,S] (transposed!)  (V)
//   mode 2: bias,       write f32 row-major     (output projection)
// block = 256 threads (8 waves). block tile: M=128 x N=128 (one head wide).
// W k-slices (128x32 bf16 = 8KB) stream through LDS via TDM, double-buffered.
// ---------------------------------------------------------------------------
__global__ __launch_bounds__(256)
void gemm_bf16_kernel(const unsigned short* __restrict__ X,
                      const unsigned short* __restrict__ W,
                      const float* __restrict__ bias,
                      void* __restrict__ outp,
                      int K, int mode) {
    __shared__ __align__(128) unsigned short wtile[2][128 * 32];   // 2 x 8KB

    const int lane = threadIdx.x & 31;
    const int w    = threadIdx.x >> 5;
    const int l15  = lane & 15;
    const int half = lane >> 4;

    const int m0 = blockIdx.x * 128 + w * 16;
    const int n0 = blockIdx.y * 128;

    v8f acc[8];
#pragma unroll
    for (int t = 0; t < 8; ++t) acc[t] = {};

    const unsigned short* xrow  = X + (size_t)(m0 + l15) * K;
    const unsigned short* wbase = W + (size_t)n0 * K;   // 128 rows of length K

    const int NKT = K / 32;
    if (w == 0) {
        tdm_load_tile_2d(wbase, (unsigned)(uintptr_t)&wtile[0][0],
                         (unsigned)K, 128u, 32u, 128u, (unsigned)K);
    }

    for (int i = 0; i < NKT; ++i) {
        const int k0 = i * 32;
        if (w == 0) {
            if (i + 1 < NKT) {
                tdm_load_tile_2d(wbase + (k0 + 32),
                                 (unsigned)(uintptr_t)&wtile[(i + 1) & 1][0],
                                 (unsigned)K, 128u, 32u, 128u, (unsigned)K);
                __builtin_amdgcn_s_wait_tensorcnt(1);   // oldest (current buf) done
            } else {
                __builtin_amdgcn_s_wait_tensorcnt(0);
            }
        }
        __syncthreads();

        Frag16 a;
        a.q[0] = *(const uint4*)(xrow + k0 + 16 * half);
        a.q[1] = *(const uint4*)(xrow + k0 + 16 * half + 8);

        const unsigned short* bt = &wtile[i & 1][0];
#pragma unroll
        for (int t = 0; t < 8; ++t) {
            Frag16 b;
            const unsigned short* br = bt + (t * 16 + l15) * 32 + 16 * half;
            b.q[0] = *(const uint4*)(br);
            b.q[1] = *(const uint4*)(br + 8);
            acc[t] = __builtin_amdgcn_wmma_f32_16x16x32_bf16(
                false, a.v, false, b.v, (short)0, acc[t], false, false);
        }
        __syncthreads();
    }

    if (mode == 2) {
        float* O = (float*)outp;
#pragma unroll
        for (int t = 0; t < 8; ++t) {
            int n = n0 + t * 16 + l15;
            float bn = bias[n];
#pragma unroll
            for (int e = 0; e < 8; ++e) {
                int m = m0 + e + 8 * half;
                O[(size_t)m * HIDDEN_ + n] = acc[t][e] + bn;
            }
        }
    } else if (mode == 1) {
        // V stored TRANSPOSED: [B,H,D,S] so attention PV B-frags are contiguous
        unsigned short* O = (unsigned short*)outp;
        int hh = n0 >> 7;
#pragma unroll
        for (int t = 0; t < 8; ++t) {
            int d = t * 16 + l15;
            float bn = bias[n0 + d];
#pragma unroll
            for (int e = 0; e < 8; ++e) {
                int m  = m0 + e + 8 * half;
                int bb = m >> 11;
                int s  = m & (S_LEN - 1);
                size_t idx = ((size_t)(bb * NHEAD + hh) * HDIM + d) * S_LEN + s;
                O[idx] = f2bf(acc[t][e] + bn);
            }
        }
    } else {  // mode 0: RoPE
        unsigned short* O = (unsigned short*)outp;   // [B,H,S,D]
        int hh = n0 >> 7;
#pragma unroll
        for (int t = 0; t < 4; ++t) {
            int d = t * 16 + l15;                 // 0..63
            float invf = __powf(10000.0f, -(float)d * (1.0f / 64.0f));
            float b1 = bias[n0 + d];
            float b2 = bias[n0 + d + 64];
#pragma unroll
            for (int e = 0; e < 8; ++e) {
                int m  = m0 + e + 8 * half;
                int bb = m >> 11;
                int s  = m & (S_LEN - 1);
                float sn, cs;
                __sincosf((float)s * invf, &sn, &cs);
                float x1 = acc[t][e] + b1;
                float x2 = acc[t + 4][e] + b2;
                size_t base = ((size_t)(bb * NHEAD + hh) * S_LEN + s) * HDIM;
                O[base + d]      = f2bf(x1 * cs - x2 * sn);
                O[base + d + 64] = f2bf(x2 * cs + x1 * sn);
            }
        }
    }
}

// ---------------------------------------------------------------------------
// Flash attention (causal). block = 256 threads (8 waves).
// wave -> 16 query rows; key tiles of 32; Q*K^T and P*V via WMMA.
// Q,K are [B,H,S,D]; V is [B,H,D,S] (transposed) so every WMMA B-fragment
// (QK^T and PV alike) is a single 32-byte contiguous per-lane read.
// ---------------------------------------------------------------------------
__global__ __launch_bounds__(256)
void attn_kernel(const unsigned short* __restrict__ Q,
                 const unsigned short* __restrict__ Kb,
                 const unsigned short* __restrict__ Vt,
                 unsigned short* __restrict__ ctx) {
    __shared__ unsigned short plds[8 * 16 * 32];   // per-wave 16x32 P tile

    const int lane = threadIdx.x & 31;
    const int w    = threadIdx.x >> 5;
    const int l15  = lane & 15;
    const int half = lane >> 4;

    const int bh = blockIdx.y;          // 0..31
    const int bb = bh >> 4;
    const int hh = bh & 15;
    const int q0 = blockIdx.x * 128 + w * 16;

    const unsigned short* qh = Q  + (size_t)bh * S_LEN * HDIM;
    const unsigned short* kh = Kb + (size_t)bh * S_LEN * HDIM;
    const unsigned short* vh = Vt + (size_t)bh * HDIM * S_LEN;  // [D][S]

    // Q fragments: 4 slices of K-dim 32
    Frag16 qa[4];
    const unsigned short* qrow = qh + (size_t)(q0 + l15) * HDIM;
#pragma unroll
    for (int dk = 0; dk < 4; ++dk) {
        qa[dk].q[0] = *(const uint4*)(qrow + dk * 32 + 16 * half);
        qa[dk].q[1] = *(const uint4*)(qrow + dk * 32 + 16 * half + 8);
    }

    v8f o[8];
#pragma unroll
    for (int t = 0; t < 8; ++t) o[t] = {};
    float mx[8], ls[8];
#pragma unroll
    for (int e = 0; e < 8; ++e) { mx[e] = -1e30f; ls[e] = 0.0f; }

    unsigned short* pw = plds + w * 512;
    const float sc = 0.08838834764831845f;        // 1/sqrt(128)

    const int ntiles = (q0 + 16 + 31) >> 5;
    for (int kt = 0; kt < ntiles; ++kt) {
        const int key0 = kt * 32;
        v8f s0 = {}, s1 = {};
#pragma unroll
        for (int dk = 0; dk < 4; ++dk) {
            Frag16 bk0, bk1;
            const unsigned short* kr0 =
                kh + (size_t)(key0 + l15) * HDIM + dk * 32 + 16 * half;
            const unsigned short* kr1 =
                kh + (size_t)(key0 + 16 + l15) * HDIM + dk * 32 + 16 * half;
            bk0.q[0] = *(const uint4*)(kr0);
            bk0.q[1] = *(const uint4*)(kr0 + 8);
            bk1.q[0] = *(const uint4*)(kr1);
            bk1.q[1] = *(const uint4*)(kr1 + 8);
            s0 = __builtin_amdgcn_wmma_f32_16x16x32_bf16(
                false, qa[dk].v, false, bk0.v, (short)0, s0, false, false);
            s1 = __builtin_amdgcn_wmma_f32_16x16x32_bf16(
                false, qa[dk].v, false, bk1.v, (short)0, s1, false, false);
        }
        // online softmax, rows m = e + 8*half live within a 16-lane half
#pragma unroll
        for (int e = 0; e < 8; ++e) {
            int m = q0 + e + 8 * half;
            float f0 = s0[e] * sc; if (key0 + l15 > m)      f0 = -1e30f;
            float f1 = s1[e] * sc; if (key0 + 16 + l15 > m) f1 = -1e30f;
            float fm = fmaxf(f0, f1);
#pragma unroll
            for (int off = 1; off < 16; off <<= 1)
                fm = fmaxf(fm, __shfl_xor(fm, off, 16));
            float mnew = fmaxf(mx[e], fm);
            float corr = __expf(mx[e] - mnew);
            float p0 = __expf(f0 - mnew);
            float p1 = __expf(f1 - mnew);
            float ps = p0 + p1;
#pragma unroll
            for (int off = 1; off < 16; off <<= 1)
                ps += __shfl_xor(ps, off, 16);
            ls[e] = ls[e] * corr + ps;
            mx[e] = mnew;
#pragma unroll
            for (int t = 0; t < 8; ++t) o[t][e] *= corr;
            // transpose P (C-frag: lane=col) into row-major LDS for A-frag
            pw[(e + 8 * half) * 32 + l15]      = f2bf(p0);
            pw[(e + 8 * half) * 32 + 16 + l15] = f2bf(p1);
        }
        asm volatile("s_wait_dscnt 0" ::: "memory");
        Frag16 pa;
        {
            const unsigned short* pr = pw + l15 * 32;
            pa.q[0] = *(const uint4*)(pr + 16 * half);
            pa.q[1] = *(const uint4*)(pr + 16 * half + 8);
        }
#pragma unroll
        for (int t = 0; t < 8; ++t) {
            // B-frag from transposed V: lane = d column, elems = 16 consecutive keys
            Frag16 vb;
            const unsigned short* vr =
                vh + (size_t)(t * 16 + l15) * S_LEN + key0 + 16 * half;
            vb.q[0] = *(const uint4*)(vr);
            vb.q[1] = *(const uint4*)(vr + 8);
            o[t] = __builtin_amdgcn_wmma_f32_16x16x32_bf16(
                false, pa.v, false, vb.v, (short)0, o[t], false, false);
        }
    }

    // normalize + store context in merged (B,S,INNER) layout, bf16
#pragma unroll
    for (int e = 0; e < 8; ++e) {
        float inv = 1.0f / ls[e];
        int m = q0 + e + 8 * half;
        size_t base = ((size_t)bb * S_LEN + m) * INNER_ + hh * HDIM;
#pragma unroll
        for (int t = 0; t < 8; ++t)
            ctx[base + t * 16 + l15] = f2bf(o[t][e] * inv);
    }
}

// ---------------------------------------------------------------------------
extern "C" void kernel_launch(void* const* d_in, const int* in_sizes, int n_in,
                              void* d_out, int out_size, void* d_ws, size_t ws_size,
                              hipStream_t stream) {
    const float* hidden = (const float*)d_in[0];
    const float* Wq = (const float*)d_in[1]; const float* bq = (const float*)d_in[2];
    const float* Wk = (const float*)d_in[3]; const float* bk = (const float*)d_in[4];
    const float* Wv = (const float*)d_in[5]; const float* bv = (const float*)d_in[6];
    const float* Wo = (const float*)d_in[7]; const float* bo = (const float*)d_in[8];

    const size_t MK = (size_t)MTOT * HIDDEN_;     // 8,388,608
    const size_t WK = (size_t)INNER_ * HIDDEN_;   // 4,194,304

    unsigned short* ws  = (unsigned short*)d_ws;
    unsigned short* Xbf = ws;
    unsigned short* Wqb = Xbf + MK;
    unsigned short* Wkb = Wqb + WK;
    unsigned short* Wvb = Wkb + WK;
    unsigned short* Wob = Wvb + WK;
    unsigned short* Qb  = Wob + WK;
    unsigned short* Kbf = Qb  + MK;
    unsigned short* Vtb = Kbf + MK;
    unsigned short* Cb  = Vtb + MK;

    f32_to_bf16_kernel<<<(int)((MK + 255) / 256), 256, 0, stream>>>(hidden, Xbf, (int)MK);
    f32_to_bf16_kernel<<<(int)((WK + 255) / 256), 256, 0, stream>>>(Wq, Wqb, (int)WK);
    f32_to_bf16_kernel<<<(int)((WK + 255) / 256), 256, 0, stream>>>(Wk, Wkb, (int)WK);
    f32_to_bf16_kernel<<<(int)((WK + 255) / 256), 256, 0, stream>>>(Wv, Wvb, (int)WK);
    f32_to_bf16_kernel<<<(int)((WK + 255) / 256), 256, 0, stream>>>(Wo, Wob, (int)WK);

    dim3 gg(MTOT / 128, INNER_ / 128);   // (32, 16)
    gemm_bf16_kernel<<<gg, 256, 0, stream>>>(Xbf, Wqb, bq, Qb,  HIDDEN_, 0);
    gemm_bf16_kernel<<<gg, 256, 0, stream>>>(Xbf, Wkb, bk, Kbf, HIDDEN_, 0);
    gemm_bf16_kernel<<<gg, 256, 0, stream>>>(Xbf, Wvb, bv, Vtb, HIDDEN_, 1);

    attn_kernel<<<dim3(S_LEN / 128, BATCH * NHEAD), 256, 0, stream>>>(Qb, Kbf, Vtb, Cb);

    gemm_bf16_kernel<<<gg, 256, 0, stream>>>(Cb, Wob, bo, d_out, HIDDEN_, 2);
}
